// MNISTBinarizedModel_1726576853663
// MI455X (gfx1250) — 8-bit matrix operand order repaired and checked by reading, compiled, not run on hardware
//
#include <hip/hip_runtime.h>

typedef __attribute__((ext_vector_type(8))) int v8i;

#define B_ROWS  65536
#define K0_REAL 784
#define K0_PAD  832
#define NH      256
#define N4_REAL 10
#define N4_PAD  16

__global__ void __launch_bounds__(256)
k_binarize_input(const float* __restrict__ x, char* __restrict__ A0, long long total)
{
    long long i = (long long)blockIdx.x * 256 + threadIdx.x;
    if (i >= total) return;
    int row = (int)(i / K0_PAD);
    int k   = (int)(i % K0_PAD);
    char v = 0;
    if (k < K0_REAL)
        v = (x[(long long)row * K0_REAL + k] >= 0.5f) ? (char)1 : (char)-1;
    A0[i] = v;
}

__global__ void __launch_bounds__(256)
k_binarize_weights(const float* __restrict__ w, char* __restrict__ wb,
                   int nreal, int kreal, int kpad, int total)
{
    int i = blockIdx.x * 256 + threadIdx.x;
    if (i >= total) return;
    int n = i / kpad;
    int k = i % kpad;
    char v = 0;
    if (n < nreal && k < kreal)
        v = (w[(long long)n * kreal + k] >= 0.0f) ? (char)1 : (char)-1;
    wb[i] = v;
}

__global__ void __launch_bounds__(256)
k_bgemm_wmma(const char* __restrict__ A, const char* __restrict__ Wb,
             short* __restrict__ Y, int N, int K)
{
    const int lane  = threadIdx.x & 31;
    const int wave  = threadIdx.x >> 5;
    const int tileM = blockIdx.x;
    const int tileN = blockIdx.y * 8 + wave;
    if (tileN * 16 >= N) return;

    const int h = lane >> 4;
    const int r = lane & 15;

    const char* aRow = A  + (size_t)(tileM * 16 + r) * K;
    const char* bRow = Wb + (size_t)(tileN * 16 + r) * K;

    v8i acc = {0, 0, 0, 0, 0, 0, 0, 0};

    for (int k0 = 0; k0 < K; k0 += 64) {
        v8i afrag, bfrag;
        #pragma unroll
        for (int j = 0; j < 8; ++j) {
            int ka = k0 + ((j >> 1) << 4) + ((j & 1) << 2) + (h << 3);
            afrag[j] = *(const int*)(aRow + ka);
        }
        #pragma unroll
        for (int j = 0; j < 8; ++j) {
            int kb = k0 + ((j >> 1) << 4) + ((j & 1) << 2) + (h << 3);
            bfrag[j] = *(const int*)(bRow + kb);
        }
        acc = __builtin_amdgcn_wmma_i32_16x16x64_iu8(
             true, afrag,  true, bfrag, acc,
             false,  false);
    }

    short* yBase = Y + (size_t)(tileM * 16) * N + tileN * 16 + r;
    #pragma unroll
    for (int j = 0; j < 8; ++j)
        yBase[(size_t)(j + 8 * h) * N] = (short)acc[j];
}

__global__ void __launch_bounds__(256)
k_zero_stats(long long* __restrict__ S, int n)
{
    int i = blockIdx.x * 256 + threadIdx.x;
    if (i < n) S[i] = 0;
}

__global__ void k_bn_stats(const short* __restrict__ Y,
                           long long* __restrict__ S1, long long* __restrict__ S2,
                           int ld, int ncols, int rowsPerBlock)
{
    int col = threadIdx.x;
    if (col >= ncols) return;
    long long s1 = 0, s2 = 0;
    int r0 = blockIdx.x * rowsPerBlock;
    for (int r = 0; r < rowsPerBlock; ++r) {
        int y = Y[(size_t)(r0 + r) * ld + col];
        s1 += y;
        s2 += (long long)y * y;
    }
    atomicAdd((unsigned long long*)(S1 + col), (unsigned long long)s1);
    atomicAdd((unsigned long long*)(S2 + col), (unsigned long long)s2);
}

__global__ void k_bn_finalize(const long long* __restrict__ S1, const long long* __restrict__ S2,
                              const float* __restrict__ gamma, const float* __restrict__ beta,
                              float* __restrict__ scale, float* __restrict__ bias, int ncols)
{
    int n = threadIdx.x;
    if (n >= ncols) return;
    const float invB = 1.0f / (float)B_ROWS;
    float mean = (float)S1[n] * invB;
    float var  = (float)S2[n] * invB - mean * mean;
    float sc   = gamma[n] * rsqrtf(var + 1e-5f);
    scale[n] = sc;
    bias[n]  = beta[n] - mean * sc;
}

__global__ void __launch_bounds__(256)
k_bn_binarize(const short* __restrict__ Y, const float* __restrict__ scale,
              const float* __restrict__ bias, char* __restrict__ Anext, long long total)
{
    long long i = (long long)blockIdx.x * 256 + threadIdx.x;
    if (i >= total) return;
    int col = (int)(i & (NH - 1));
    float z = fmaf(scale[col], (float)Y[i], bias[col]);
    Anext[i] = (z >= 0.0f) ? (char)1 : (char)-1;
}

__global__ void __launch_bounds__(256)
k_softmax_col(const short* __restrict__ Y4, const float* __restrict__ scale,
              const float* __restrict__ bias, float* __restrict__ out)
{
    const int col = blockIdx.x;
    const int tid = threadIdx.x;
    const float sc = scale[col], bi = bias[col];
    __shared__ float red[256];

    float m = -3.4e38f;
    for (int r = tid; r < B_ROWS; r += 256)
        m = fmaxf(m, fmaf(sc, (float)Y4[(size_t)r * N4_PAD + col], bi));
    red[tid] = m; __syncthreads();
    for (int s = 128; s > 0; s >>= 1) {
        if (tid < s) red[tid] = fmaxf(red[tid], red[tid + s]);
        __syncthreads();
    }
    m = red[0]; __syncthreads();

    float ssum = 0.0f;
    for (int r = tid; r < B_ROWS; r += 256)
        ssum += expf(fmaf(sc, (float)Y4[(size_t)r * N4_PAD + col], bi) - m);
    red[tid] = ssum; __syncthreads();
    for (int s = 128; s > 0; s >>= 1) {
        if (tid < s) red[tid] += red[tid + s];
        __syncthreads();
    }
    float denom = red[0];

    for (int r = tid; r < B_ROWS; r += 256) {
        float z = fmaf(sc, (float)Y4[(size_t)r * N4_PAD + col], bi);
        out[(size_t)r * N4_REAL + col] = expf(z - m) / denom;
    }
}

extern "C" void kernel_launch(void* const* d_in, const int* in_sizes, int n_in,
                              void* d_out, int out_size, void* d_ws, size_t ws_size,
                              hipStream_t stream)
{
    const float* x = (const float*)d_in[0];
    const float* w[5];  const float* gamma[5];  const float* beta[5];
    for (int i = 0; i < 5; ++i) {
        w[i]     = (const float*)d_in[1 + 3 * i];
        gamma[i] = (const float*)d_in[2 + 3 * i];
        beta[i]  = (const float*)d_in[3 + 3 * i];
    }

    char* ws = (char*)d_ws;
    size_t off = 0;
    char* A0  = ws + off; off += (size_t)B_ROWS * K0_PAD;
    char* Wb0 = ws + off; off += (size_t)NH * K0_PAD;
    char* Wb1 = ws + off; off += (size_t)NH * NH;
    char* Wb2 = ws + off; off += (size_t)NH * NH;
    char* Wb3 = ws + off; off += (size_t)NH * NH;
    char* Wb4 = ws + off; off += (size_t)N4_PAD * NH;
    char* Aping = ws + off; off += (size_t)B_ROWS * NH;
    char* Apong = ws + off; off += (size_t)B_ROWS * NH;
    short* Y  = (short*)(ws + off); off += (size_t)B_ROWS * NH * 2;
    long long* S1 = (long long*)(ws + off); off += (size_t)NH * 8;
    long long* S2 = (long long*)(ws + off); off += (size_t)NH * 8;
    float* scale  = (float*)(ws + off);    off += (size_t)NH * 4;
    float* bias   = (float*)(ws + off);    off += (size_t)NH * 4;

    {
        long long total = (long long)B_ROWS * K0_PAD;
        k_binarize_input<<<(int)((total + 255) / 256), 256, 0, stream>>>(x, A0, total);
    }

    k_binarize_weights<<<(NH * K0_PAD + 255) / 256, 256, 0, stream>>>(w[0], Wb0, NH, K0_REAL, K0_PAD, NH * K0_PAD);
    k_binarize_weights<<<(NH * NH + 255) / 256, 256, 0, stream>>>(w[1], Wb1, NH, NH, NH, NH * NH);
    k_binarize_weights<<<(NH * NH + 255) / 256, 256, 0, stream>>>(w[2], Wb2, NH, NH, NH, NH * NH);
    k_binarize_weights<<<(NH * NH + 255) / 256, 256, 0, stream>>>(w[3], Wb3, NH, NH, NH, NH * NH);
    k_binarize_weights<<<(N4_PAD * NH + 255) / 256, 256, 0, stream>>>(w[4], Wb4, N4_REAL, NH, NH, N4_PAD * NH);

    char* Wbs[5]  = {Wb0, Wb1, Wb2, Wb3, Wb4};
    char* Abuf[2] = {Aping, Apong};
    const char* cur = A0;
    int curK = K0_PAD;

    for (int k = 0; k < 5; ++k) {
        const int Np    = (k < 4) ? NH : N4_PAD;
        const int ncols = (k < 4) ? NH : N4_REAL;

        dim3 grid(B_ROWS / 16, (Np + 127) / 128);
        k_bgemm_wmma<<<grid, 256, 0, stream>>>(cur, Wbs[k], Y, Np, curK);

        k_zero_stats<<<2, 256, 0, stream>>>(S1, 2 * NH);
        k_bn_stats<<<256, Np, 0, stream>>>(Y, S1, S2, Np, ncols, B_ROWS / 256);
        k_bn_finalize<<<1, 256, 0, stream>>>(S1, S2, gamma[k], beta[k], scale, bias, ncols);

        if (k < 4) {
            long long total = (long long)B_ROWS * NH;
            k_bn_binarize<<<(int)((total + 255) / 256), 256, 0, stream>>>(Y, scale, bias, Abuf[k & 1], total);
            cur  = Abuf[k & 1];
            curK = NH;
        }
    }

    k_softmax_col<<<N4_REAL, 256, 0, stream>>>(Y, scale, bias, (float*)d_out);
}
